// HybridEquivariantGNN_35029753266205
// MI455X (gfx1250) — matrix operand order repaired and checked by reading, not compiled, not run on hardware
//
#include <hip/hip_runtime.h>
#include <hip/hip_bf16.h>
#include <stdint.h>

// ---------------------------------------------------------------------------
// CDNA5 (gfx1250) WMMA types and helpers
// ---------------------------------------------------------------------------
typedef __bf16        bf16x16 __attribute__((ext_vector_type(16)));
typedef float         f32x8   __attribute__((ext_vector_type(8)));
typedef float         f32x4v  __attribute__((ext_vector_type(4)));
typedef unsigned int  u32x4v  __attribute__((ext_vector_type(4)));

union Frag {
  bf16x16        v;
  unsigned short u[16];
  u32x4v         q[2];
};

__device__ __forceinline__ unsigned short f2bf(float f) {
  unsigned int u = __builtin_bit_cast(unsigned int, f);
  u += 0x7fffu + ((u >> 16) & 1u);            // round-to-nearest-even
  return (unsigned short)(u >> 16);
}
__device__ __forceinline__ float bf2f(unsigned short h) {
  return __builtin_bit_cast(float, ((unsigned int)h) << 16);
}
__device__ __forceinline__ float silu(float x) {
  return x / (1.0f + __expf(-x));
}

#define WMMA_BF16(A, B, C) \
  __builtin_amdgcn_wmma_f32_16x16x32_bf16(false, (A), false, (B), (short)0, (C), false, false)

#define WPB 4  // waves (16-row tiles) per block

// ---------------------------------------------------------------------------
// Pack an fp32 [K x 64] weight matrix into bf16 WMMA B-fragments.
// B-fragment layout (16x16x32 bf16): lane = (kgrp<<4)|n, element e -> K = (e>>3)*16+kgrp*8+(e&7).
// Storage: frag[((kt*4 + nt)*32 + lane)*16 + e]
// ---------------------------------------------------------------------------
__global__ void k_pack_b(const float* __restrict__ w, int Kreal, int KT,
                         unsigned short* __restrict__ out) {
  int t = blockIdx.x * blockDim.x + threadIdx.x;
  if (t >= KT * 128) return;
  int lane = t & 31;
  int nt   = (t >> 5) & 3;
  (void)nt;
  int kt   = t >> 7;
  unsigned short* dst = out + (size_t)t * 16;
  int n     = ((t >> 5) & 3) * 16 + (lane & 15);
  int kbase = kt * 32 + (lane >> 4) * 8;
#pragma unroll
  for (int e = 0; e < 16; ++e) {
    int k = kbase + ((e >> 3) << 4) + (e & 7);
    float v = (k < Kreal) ? w[(size_t)k * 64 + n] : 0.0f;
    dst[e] = f2bf(v);
  }
}

__global__ void k_zero(float* __restrict__ p, long long n) {
  long long i = (long long)blockIdx.x * blockDim.x + threadIdx.x;
  if (i < n) p[i] = 0.0f;
}

__global__ void k_node_init(const int* __restrict__ ntypes, const float* __restrict__ emb,
                            const float* __restrict__ pos, float* __restrict__ h32,
                            unsigned short* __restrict__ hbf, float* __restrict__ pos_cur,
                            int N) {
  int i = blockIdx.x * blockDim.x + threadIdx.x;
  if (i >= N) return;
  int t = ntypes[i];
#pragma unroll 8
  for (int c = 0; c < 64; ++c) {
    float v = emb[(size_t)t * 64 + c];
    h32[(size_t)i * 64 + c] = v;
    hbf[(size_t)i * 64 + c] = f2bf(v);
  }
  pos_cur[i * 3 + 0] = pos[i * 3 + 0];
  pos_cur[i * 3 + 1] = pos[i * 3 + 1];
  pos_cur[i * 3 + 2] = pos[i * 3 + 2];
}

__global__ void k_deg(const int* __restrict__ eidx, int E, float* __restrict__ deg) {
  int e = blockIdx.x * blockDim.x + threadIdx.x;
  if (e < E) atomicAdd(&deg[eidx[E + e]], 1.0f);
}

// ---------------------------------------------------------------------------
// Edge kernel: one wave32 per 16-edge tile.
//   m1 = silu([h[d]|h[s]|d2|ea] @ ew1 + eb1)           (K = 160 padded, WMMA)
//   m  = silu(m1 @ ew2 + eb2)                          (WMMA)
//   agg[dst]  += m                                     (L2 float atomics)
//   c  = silu(m @ cw1 + cb1) @ cw2 + cb2               (WMMA + dot)
//   pacc[dst] += rel * c
// ---------------------------------------------------------------------------
__global__ __launch_bounds__(WPB * 32) void k_edge(
    const int* __restrict__ eidx, int E,
    const float* __restrict__ pos, const float* __restrict__ eattr,
    const unsigned short* __restrict__ hbf,
    const unsigned short* __restrict__ ew1f, const float* __restrict__ eb1,
    const unsigned short* __restrict__ ew2f, const float* __restrict__ eb2,
    const unsigned short* __restrict__ cw1f, const float* __restrict__ cb1,
    const float* __restrict__ cw2, const float* __restrict__ cb2,
    float* __restrict__ agg, float* __restrict__ pacc, int ntiles) {
  __shared__ unsigned short lds_m[WPB][16 * 64];
  __shared__ unsigned short lds_t[WPB][16 * 64];
  __shared__ int            lds_d[WPB][16];

  const int wave = threadIdx.x >> 5;
  const int lane = threadIdx.x & 31;
  const int tile = blockIdx.x * WPB + wave;
  if (tile >= ntiles) return;                      // uniform per wave
  const int row = lane & 15;
  const int hi  = lane >> 4;

  const int  edge   = tile * 16 + row;
  const bool evalid = edge < E;
  const int  ec     = evalid ? edge : 0;

  const int s = eidx[ec];
  const int d = eidx[E + ec];
  if (hi == 0) lds_d[wave][row] = evalid ? d : -1;

  const float rx = pos[d * 3 + 0] - pos[s * 3 + 0];
  const float ry = pos[d * 3 + 1] - pos[s * 3 + 1];
  const float rz = pos[d * 3 + 2] - pos[s * 3 + 2];
  const float d2 = rx * rx + ry * ry + rz * rz;
  const float ea0 = eattr[(size_t)ec * 4 + 0];
  const float ea1 = eattr[(size_t)ec * 4 + 1];
  const float ea2 = eattr[(size_t)ec * 4 + 2];
  const float ea3 = eattr[(size_t)ec * 4 + 3];

  // ---- GEMM 1: m_in[16x160] @ ew1[160x64] ----
  f32x8 acc[4] = {};
#pragma unroll
  for (int kt = 0; kt < 5; ++kt) {
    Frag a;
    if (kt < 4) {
      const unsigned short* base =
          hbf + (size_t)(kt < 2 ? d : s) * 64 + (kt & 1) * 32 + hi * 8;
      a.q[0] = *(const u32x4v*)(base);
      a.q[1] = *(const u32x4v*)(base + 16);
    } else {
#pragma unroll
      for (int e = 0; e < 16; ++e) {
        int k = 128 + ((e >> 3) << 4) + hi * 8 + (e & 7);
        float v = 0.0f;
        if (k == 128) v = d2;
        else if (k == 129) v = ea0;
        else if (k == 130) v = ea1;
        else if (k == 131) v = ea2;
        else if (k == 132) v = ea3;
        a.u[e] = f2bf(v);
      }
    }
#pragma unroll
    for (int nt = 0; nt < 4; ++nt) {
      Frag b;
      const unsigned short* bp = ew1f + ((size_t)((kt * 4 + nt) * 32 + lane)) * 16;
      b.q[0] = *(const u32x4v*)(bp);
      b.q[1] = *(const u32x4v*)(bp + 8);
      acc[nt] = WMMA_BF16(a.v, b.v, acc[nt]);
    }
  }
#pragma unroll
  for (int nt = 0; nt < 4; ++nt) {
    const int col  = nt * 16 + row;
    const float bi = eb1[col];
#pragma unroll
    for (int r = 0; r < 8; ++r) {
      float v = silu(acc[nt][r] + bi);
      lds_m[wave][(r + hi * 8) * 64 + col] = f2bf(v);
    }
  }
  __builtin_amdgcn_wave_barrier();

  // ---- GEMM 2: m1[16x64] @ ew2[64x64] ----
  f32x8 acc2[4] = {};
#pragma unroll
  for (int kt = 0; kt < 2; ++kt) {
    Frag a;
    const unsigned short* mb = &lds_m[wave][row * 64 + kt * 32 + hi * 8];
    a.q[0] = *(const u32x4v*)(mb);
    a.q[1] = *(const u32x4v*)(mb + 16);
#pragma unroll
    for (int nt = 0; nt < 4; ++nt) {
      Frag b;
      const unsigned short* bp = ew2f + ((size_t)((kt * 4 + nt) * 32 + lane)) * 16;
      b.q[0] = *(const u32x4v*)(bp);
      b.q[1] = *(const u32x4v*)(bp + 8);
      acc2[nt] = WMMA_BF16(a.v, b.v, acc2[nt]);
    }
  }
  __builtin_amdgcn_wave_barrier();

  // m = silu(. + eb2): scatter to agg, stash for coord MLP
#pragma unroll
  for (int nt = 0; nt < 4; ++nt) {
    const int col  = nt * 16 + row;
    const float bi = eb2[col];
#pragma unroll
    for (int r = 0; r < 8; ++r) {
      float v = silu(acc2[nt][r] + bi);
      lds_m[wave][(r + hi * 8) * 64 + col] = f2bf(v);
      const int dr = lds_d[wave][r + hi * 8];
      if (dr >= 0) atomicAdd(&agg[(size_t)dr * 64 + col], v);
    }
  }
  __builtin_amdgcn_wave_barrier();

  // ---- coord MLP: silu(m @ cw1 + cb1) ----
  f32x8 acc3[4] = {};
#pragma unroll
  for (int kt = 0; kt < 2; ++kt) {
    Frag a;
    const unsigned short* mb = &lds_m[wave][row * 64 + kt * 32 + hi * 8];
    a.q[0] = *(const u32x4v*)(mb);
    a.q[1] = *(const u32x4v*)(mb + 16);
#pragma unroll
    for (int nt = 0; nt < 4; ++nt) {
      Frag b;
      const unsigned short* bp = cw1f + ((size_t)((kt * 4 + nt) * 32 + lane)) * 16;
      b.q[0] = *(const u32x4v*)(bp);
      b.q[1] = *(const u32x4v*)(bp + 8);
      acc3[nt] = WMMA_BF16(a.v, b.v, acc3[nt]);
    }
  }
#pragma unroll
  for (int nt = 0; nt < 4; ++nt) {
    const int col  = nt * 16 + row;
    const float bi = cb1[col];
#pragma unroll
    for (int r = 0; r < 8; ++r)
      lds_t[wave][(r + hi * 8) * 64 + col] = f2bf(silu(acc3[nt][r] + bi));
  }
  __builtin_amdgcn_wave_barrier();

  if (hi == 0 && evalid) {
    float c = cb2[0];
#pragma unroll 8
    for (int k = 0; k < 64; ++k)
      c += bf2f(lds_t[wave][row * 64 + k]) * cw2[k];
    atomicAdd(&pacc[(size_t)d * 3 + 0], rx * c);
    atomicAdd(&pacc[(size_t)d * 3 + 1], ry * c);
    atomicAdd(&pacc[(size_t)d * 3 + 2], rz * c);
  }
}

// ---------------------------------------------------------------------------
// Node kernel: h += silu([h|agg] @ nw1 + nb1) @ nw2 + nb2 ; pos += pacc/deg ;
// re-zero agg/pacc for next layer.  One wave32 per 16-node tile.
// ---------------------------------------------------------------------------
__global__ __launch_bounds__(WPB * 32) void k_node(
    float* __restrict__ h32, unsigned short* __restrict__ hbf,
    float* __restrict__ agg, float* __restrict__ pacc,
    float* __restrict__ pos_cur, const float* __restrict__ deg,
    const unsigned short* __restrict__ nw1f, const float* __restrict__ nb1,
    const unsigned short* __restrict__ nw2f, const float* __restrict__ nb2,
    int N, int ntiles) {
  __shared__ unsigned short lds_s[WPB][16 * 64];

  const int wave = threadIdx.x >> 5;
  const int lane = threadIdx.x & 31;
  const int tile = blockIdx.x * WPB + wave;
  if (tile >= ntiles) return;
  const int row = lane & 15;
  const int hi  = lane >> 4;

  const int  node  = tile * 16 + row;
  const bool nvld  = node < N;
  const int  nodec = nvld ? node : 0;

  f32x8 acc[4] = {};
#pragma unroll
  for (int kt = 0; kt < 4; ++kt) {
    Frag a;
    if (kt < 2) {
      const unsigned short* base = hbf + (size_t)nodec * 64 + kt * 32 + hi * 8;
      a.q[0] = *(const u32x4v*)(base);
      a.q[1] = *(const u32x4v*)(base + 16);
    } else {
      const float* ab = agg + (size_t)nodec * 64 + (kt - 2) * 32 + hi * 8;
      f32x4v f0 = *(const f32x4v*)(ab);
      f32x4v f1 = *(const f32x4v*)(ab + 4);
      f32x4v f2 = *(const f32x4v*)(ab + 16);
      f32x4v f3 = *(const f32x4v*)(ab + 20);
#pragma unroll
      for (int j = 0; j < 4; ++j) {
        a.u[j]      = f2bf(f0[j]);
        a.u[4 + j]  = f2bf(f1[j]);
        a.u[8 + j]  = f2bf(f2[j]);
        a.u[12 + j] = f2bf(f3[j]);
      }
    }
#pragma unroll
    for (int nt = 0; nt < 4; ++nt) {
      Frag b;
      const unsigned short* bp = nw1f + ((size_t)((kt * 4 + nt) * 32 + lane)) * 16;
      b.q[0] = *(const u32x4v*)(bp);
      b.q[1] = *(const u32x4v*)(bp + 8);
      acc[nt] = WMMA_BF16(a.v, b.v, acc[nt]);
    }
  }
#pragma unroll
  for (int nt = 0; nt < 4; ++nt) {
    const int col  = nt * 16 + row;
    const float bi = nb1[col];
#pragma unroll
    for (int r = 0; r < 8; ++r)
      lds_s[wave][(r + hi * 8) * 64 + col] = f2bf(silu(acc[nt][r] + bi));
  }
  __builtin_amdgcn_wave_barrier();

  f32x8 acc2[4] = {};
#pragma unroll
  for (int kt = 0; kt < 2; ++kt) {
    Frag a;
    const unsigned short* mb = &lds_s[wave][row * 64 + kt * 32 + hi * 8];
    a.q[0] = *(const u32x4v*)(mb);
    a.q[1] = *(const u32x4v*)(mb + 16);
#pragma unroll
    for (int nt = 0; nt < 4; ++nt) {
      Frag b;
      const unsigned short* bp = nw2f + ((size_t)((kt * 4 + nt) * 32 + lane)) * 16;
      b.q[0] = *(const u32x4v*)(bp);
      b.q[1] = *(const u32x4v*)(bp + 8);
      acc2[nt] = WMMA_BF16(a.v, b.v, acc2[nt]);
    }
  }

  // h update (D-layout: lane owns column col, rows r + 8*hi of this tile)
#pragma unroll
  for (int nt = 0; nt < 4; ++nt) {
    const int col = nt * 16 + row;
#pragma unroll
    for (int r = 0; r < 8; ++r) {
      const int n2 = tile * 16 + r + hi * 8;
      if (n2 < N) {
        const size_t off = (size_t)n2 * 64 + col;
        float hv = h32[off] + acc2[nt][r] + nb2[col];
        h32[off] = hv;
        hbf[off] = f2bf(hv);
      }
    }
  }

  if (nvld) {
    // zero this node's agg half (lane pair covers 64 channels)
    float* az = agg + (size_t)node * 64 + hi * 32;
    f32x4v z = {0.0f, 0.0f, 0.0f, 0.0f};
#pragma unroll
    for (int j = 0; j < 8; ++j) *(f32x4v*)(az + j * 4) = z;
    if (hi == 0) {
      float dg = fmaxf(deg[node], 1.0f);
#pragma unroll
      for (int c = 0; c < 3; ++c) {
        pos_cur[node * 3 + c] += pacc[node * 3 + c] / dg;
        pacc[node * 3 + c] = 0.0f;
      }
    }
  }
}

__global__ void k_pool(const float* __restrict__ h32, const int* __restrict__ batch,
                       float* __restrict__ gsum, long long total) {
  long long i = (long long)blockIdx.x * blockDim.x + threadIdx.x;
  if (i >= total) return;
  int node = (int)(i >> 6);
  int col  = (int)(i & 63);
  atomicAdd(&gsum[(size_t)batch[node] * 64 + col], h32[i]);
}

__global__ void k_cnt(const int* __restrict__ batch, float* __restrict__ gcnt, int N) {
  int i = blockIdx.x * blockDim.x + threadIdx.x;
  if (i < N) atomicAdd(&gcnt[batch[i]], 1.0f);
}

// ---------------------------------------------------------------------------
// Pool head: g -> q_in -> 4-qubit statevector circuit -> post-MLP. 1 thread/graph.
// ---------------------------------------------------------------------------
__global__ __launch_bounds__(128) void k_final(
    const float* __restrict__ gsum, const float* __restrict__ gcnt,
    const float* __restrict__ prew, const float* __restrict__ preb,
    const float* __restrict__ qw,
    const float* __restrict__ pw1, const float* __restrict__ pb1,
    const float* __restrict__ pw2, const float* __restrict__ pb2,
    float* __restrict__ out, int G) {
  int g = threadIdx.x;
  if (g >= G) return;
  float cnt = fmaxf(gcnt[g], 1.0f);
  float gv[64];
#pragma unroll 8
  for (int k = 0; k < 64; ++k) gv[k] = gsum[(size_t)g * 64 + k] / cnt;

  float x[4];
#pragma unroll
  for (int j = 0; j < 4; ++j) {
    float s = preb[j];
#pragma unroll 8
    for (int k = 0; k < 64; ++k) s += gv[k] * prew[k * 4 + j];
    x[j] = s;
  }

  float rr[16] = {}, ri[16] = {};
  rr[0] = 1.0f;
  // RX(x[q]) on each qubit (qubit q <-> bit 3-q)
#pragma unroll
  for (int q = 0; q < 4; ++q) {
    float c = __cosf(0.5f * x[q]), s = __sinf(0.5f * x[q]);
    int mask = 8 >> q;
#pragma unroll
    for (int i = 0; i < 16; ++i) {
      if (i & mask) continue;
      int i1 = i | mask;
      float ar = rr[i], ai = ri[i], br = rr[i1], bi = ri[i1];
      rr[i]  = c * ar + s * bi;  ri[i]  = c * ai - s * br;
      rr[i1] = c * br + s * ai;  ri[i1] = c * bi - s * ar;
    }
  }
#pragma unroll
  for (int l = 0; l < 2; ++l) {
#pragma unroll
    for (int q = 0; q < 4; ++q) {
      float t = qw[l * 4 + q];
      float c = __cosf(0.5f * t), s = __sinf(0.5f * t);
      int mask = 8 >> q;
#pragma unroll
      for (int i = 0; i < 16; ++i) {
        if (i & mask) continue;
        int i1 = i | mask;
        float ar = rr[i], ai = ri[i], br = rr[i1], bi = ri[i1];
        rr[i]  = c * ar - s * br;  ri[i]  = c * ai - s * bi;
        rr[i1] = s * ar + c * br;  ri[i1] = s * ai + c * bi;
      }
    }
#pragma unroll
    for (int q = 0; q < 4; ++q) {  // CNOT q -> (q+1)%4
      int cm = 8 >> q, tm = 8 >> ((q + 1) & 3);
#pragma unroll
      for (int i = 0; i < 16; ++i) {
        if ((i & cm) && !(i & tm)) {
          int j = i | tm;
          float tr = rr[i], ti = ri[i];
          rr[i] = rr[j]; ri[i] = ri[j];
          rr[j] = tr;    ri[j] = ti;
        }
      }
    }
  }
  float z[4];
#pragma unroll
  for (int q = 0; q < 4; ++q) {
    float s = 0.0f;
#pragma unroll
    for (int i = 0; i < 16; ++i) {
      float p = rr[i] * rr[i] + ri[i] * ri[i];
      s += ((i >> (3 - q)) & 1) ? -p : p;
    }
    z[q] = s;
  }
  float y = pb2[0];
#pragma unroll 4
  for (int j = 0; j < 64; ++j) {
    float h = pb1[j];
#pragma unroll
    for (int k = 0; k < 4; ++k) h += z[k] * pw1[k * 64 + j];
    y += silu(h) * pw2[j];
  }
  out[g] = y;
}

// ---------------------------------------------------------------------------
// Host launcher
// ---------------------------------------------------------------------------
extern "C" void kernel_launch(void* const* d_in, const int* in_sizes, int n_in,
                              void* d_out, int out_size, void* d_ws, size_t ws_size,
                              hipStream_t stream) {
  const int*   node_types = (const int*)d_in[0];
  const float* pos   = (const float*)d_in[1];
  const int*   eidx  = (const int*)d_in[2];
  const float* eattr = (const float*)d_in[3];
  const int*   batch = (const int*)d_in[4];
  const float* emb   = (const float*)d_in[5];
  const float* ew1 = (const float*)d_in[6];  const float* eb1 = (const float*)d_in[7];
  const float* ew2 = (const float*)d_in[8];  const float* eb2 = (const float*)d_in[9];
  const float* cw1 = (const float*)d_in[10]; const float* cb1 = (const float*)d_in[11];
  const float* cw2 = (const float*)d_in[12]; const float* cb2 = (const float*)d_in[13];
  const float* nw1 = (const float*)d_in[14]; const float* nb1 = (const float*)d_in[15];
  const float* nw2 = (const float*)d_in[16]; const float* nb2 = (const float*)d_in[17];
  const float* prew= (const float*)d_in[18]; const float* preb= (const float*)d_in[19];
  const float* qw  = (const float*)d_in[20];
  const float* pw1 = (const float*)d_in[21]; const float* pb1 = (const float*)d_in[22];
  const float* pw2 = (const float*)d_in[23]; const float* pb2 = (const float*)d_in[24];

  const int N = in_sizes[0];
  const int E = in_sizes[2] / 2;
  const int G = 128;

  // workspace carve-out
  char* ws = (char*)d_ws;
  size_t off = 0;
  auto give = [&](size_t bytes) -> char* {
    char* p = ws + off;
    off = (off + bytes + 255) & ~(size_t)255;
    return p;
  };
  float*          pos_cur = (float*)give((size_t)N * 3 * 4);
  float*          deg     = (float*)give((size_t)N * 4);
  float*          h32     = (float*)give((size_t)N * 64 * 4);
  unsigned short* hbf     = (unsigned short*)give((size_t)N * 64 * 2);
  float*          agg     = (float*)give((size_t)N * 64 * 4);
  float*          pacc    = (float*)give((size_t)N * 3 * 4);
  float*          gsum    = (float*)give((size_t)G * 64 * 4);
  float*          gcnt    = (float*)give((size_t)G * 4);
  unsigned short* frags   = (unsigned short*)give((size_t)3 * 30720 * 2);

  auto zf = [&](float* p, long long n) {
    k_zero<<<(int)((n + 255) / 256), 256, 0, stream>>>(p, n);
  };
  zf(deg, N);
  zf(agg, (long long)N * 64);
  zf(pacc, (long long)N * 3);
  zf(gsum, (long long)G * 64);
  zf(gcnt, G);

  k_node_init<<<(N + 127) / 128, 128, 0, stream>>>(node_types, emb, pos, h32, hbf, pos_cur, N);
  k_deg<<<(E + 255) / 256, 256, 0, stream>>>(eidx, E, deg);

  // pack all weight matrices into WMMA B-fragment order (bf16)
  for (int l = 0; l < 3; ++l) {
    unsigned short* base = frags + (size_t)l * 30720;
    k_pack_b<<<5, 128, 0, stream>>>(ew1 + (size_t)l * 133 * 64, 133, 5, base);
    k_pack_b<<<2, 128, 0, stream>>>(ew2 + (size_t)l * 64 * 64, 64, 2, base + 10240);
    k_pack_b<<<2, 128, 0, stream>>>(cw1 + (size_t)l * 64 * 64, 64, 2, base + 14336);
    k_pack_b<<<4, 128, 0, stream>>>(nw1 + (size_t)l * 128 * 64, 128, 4, base + 18432);
    k_pack_b<<<2, 128, 0, stream>>>(nw2 + (size_t)l * 64 * 64, 64, 2, base + 26624);
  }

  const int etiles = (E + 15) / 16;
  const int ntiles = (N + 15) / 16;
  for (int l = 0; l < 3; ++l) {
    unsigned short* base = frags + (size_t)l * 30720;
    k_edge<<<(etiles + WPB - 1) / WPB, WPB * 32, 0, stream>>>(
        eidx, E, pos_cur, eattr, hbf,
        base,         eb1 + l * 64,
        base + 10240, eb2 + l * 64,
        base + 14336, cb1 + l * 64,
        cw2 + (size_t)l * 64, cb2 + l,
        agg, pacc, etiles);
    k_node<<<(ntiles + WPB - 1) / WPB, WPB * 32, 0, stream>>>(
        h32, hbf, agg, pacc, pos_cur, deg,
        base + 18432, nb1 + l * 64,
        base + 26624, nb2 + l * 64,
        N, ntiles);
  }

  k_pool<<<(int)(((long long)N * 64 + 255) / 256), 256, 0, stream>>>(h32, batch, gsum,
                                                                     (long long)N * 64);
  k_cnt<<<(N + 255) / 256, 256, 0, stream>>>(batch, gcnt, N);
  k_final<<<1, 128, 0, stream>>>(gsum, gcnt, prew, preb, qw, pw1, pb1, pw2, pb2,
                                 (float*)d_out, G);
}
